// GCModel_47931835023429
// MI455X (gfx1250) — matrix operand order repaired and checked by reading, not compiled, not run on hardware
//
#include <hip/hip_runtime.h>
#include <hip/hip_bf16.h>
#include <math.h>

#define B_SZ 2048
#define EPSV 1e-05f
#define PI_F 3.14159265358979f

typedef __attribute__((ext_vector_type(16))) _Float16 v16h;
typedef __attribute__((ext_vector_type(8)))  _Float16 v8h;
typedef __attribute__((ext_vector_type(8)))  float    v8f;
typedef __attribute__((ext_vector_type(4)))  float    v4f;
typedef __attribute__((ext_vector_type(4)))  unsigned int u32x4;
typedef __attribute__((ext_vector_type(8)))  int     i32x8;

// ---------------- complex helpers ----------------
struct cpx { float x, y; };
__device__ __forceinline__ cpx cmk(float a, float b){ cpx c; c.x=a; c.y=b; return c; }
__device__ __forceinline__ cpx operator+(cpx a, cpx b){ return cmk(a.x+b.x, a.y+b.y); }
__device__ __forceinline__ cpx operator-(cpx a, cpx b){ return cmk(a.x-b.x, a.y-b.y); }
__device__ __forceinline__ cpx operator*(cpx a, cpx b){ return cmk(a.x*b.x - a.y*b.y, a.x*b.y + a.y*b.x); }
__device__ __forceinline__ cpx operator*(float s, cpx a){ return cmk(s*a.x, s*a.y); }
__device__ __forceinline__ float cabsf_(cpx a){ return sqrtf(a.x*a.x + a.y*a.y); }
__device__ __forceinline__ cpx cexp_i(float t){ float s, c; __sincosf(t, &s, &c); return cmk(c, s); }

__device__ __forceinline__ cpx csigmoid_c(cpx z){
  float e = expf(-z.x);
  float dr = 1.f + e * cosf(z.y);
  float di = -e * sinf(z.y);
  float n2 = dr*dr + di*di;
  return cmk(dr / n2, -di / n2);
}
__device__ __forceinline__ cpx ctanh_c(cpx z){
  float d = coshf(2.f*z.x) + cosf(2.f*z.y);
  return cmk(sinhf(2.f*z.x)/d, sinf(2.f*z.y)/d);
}
__device__ __forceinline__ float sigmoidf_(float x){ return 1.f/(1.f+expf(-x)); }
__device__ __forceinline__ float mishf_(float x){
  float sp = (x > 15.f) ? x : log1pf(expf(x));
  return x * tanhf(sp);
}

// k-th smallest (0-indexed) of |base[i*stride]| over n elements, O(n^2) selection
__device__ float kth_abs(const cpx* base, int n, int stride, int k){
  float r = 0.f;
  for (int i = 0; i < n; ++i){
    float ai = cabsf_(base[i*stride]);
    int lt = 0, le = 0;
    for (int j = 0; j < n; ++j){
      float aj = cabsf_(base[j*stride]);
      lt += (aj <  ai);
      le += (aj <= ai);
    }
    if (lt <= k && k < le) r = ai;
  }
  return r;
}

// ---------------- WMMA fragment helpers (contiguous k = 16*h + e per lane, wave32) ----------------
// Generic A gather (global f32, contiguous in k per lane -> vectorizes to b128):
template <typename G>
__device__ __forceinline__ v16h frag_a(int lane, G&& g) {
  int m = lane & 15, h = lane >> 4;
  v16h f;
#pragma unroll
  for (int e = 0; e < 16; ++e) {
    int v = e >> 1, pos = e & 1;
    int k = v * 2 + pos + h * 16;
    f[e] = (_Float16)g(m, k);
  }
  return f;
}
// Generic B gather (global f32, contiguous in k per lane)
template <typename G>
__device__ __forceinline__ v16h frag_b(int lane, G&& g) {
  int n = lane & 15, h = lane >> 4;
  v16h f;
#pragma unroll
  for (int e = 0; e < 16; ++e) f[e] = (_Float16)g(h * 16 + e, n);
  return f;
}
__device__ __forceinline__ v16h cat8(v8h a, v8h b){
  return __builtin_shufflevector(a, b, 0,1,2,3,4,5,6,7,8,9,10,11,12,13,14,15);
}
// B fragment from f16 LDS, N-major [n][K]
__device__ __forceinline__ v16h ldB16(const _Float16* row, int h){
  const v8h* p = (const v8h*)(row + h*16);
  return cat8(p[0], p[1]);
}
// B fragment from f32 LDS, N-major [n][K]: 4x ds_load_b128 + pack-convert
__device__ __forceinline__ v16h ldB32f(const float* row, int h){
  const v4f* p = (const v4f*)(row + h*16);
  v4f a = p[0], b = p[1], c = p[2], d = p[3];
  v16h f;
#pragma unroll
  for (int e = 0; e < 4; ++e) {
    f[e]      = (_Float16)a[e];
    f[4 + e]  = (_Float16)b[e];
    f[8 + e]  = (_Float16)c[e];
    f[12 + e] = (_Float16)d[e];
  }
  return f;
}
// A fragment from f16 LDS, M-major [m][K]
__device__ __forceinline__ v16h ldA16(const _Float16* row, int h){
  const v8h* p0 = (const v8h*)(row + h*16);
  const v8h* p1 = (const v8h*)(row + h*16 + 8);
  return cat8(*p0, *p1);
}
__device__ __forceinline__ v8f wmma_f16(v16h a, v16h b, v8f c) {
  return __builtin_amdgcn_wmma_f32_16x16x32_f16(false, a, false, b, (short)0, c, false, false);
}
// D/C (16x16 f32): lane n = lane&15, VGPR v -> row m = v + 8*(lane>>4)

// =============================================================================
// K1: front end. One block per batch element, 256 threads.
// Emits xr (B,S,F) real [conv1 B-operand order] and cxlast (B,7,F) complex.
// =============================================================================
__global__ void __launch_bounds__(256) k_frontend(
    const int* __restrict__ day, const float* __restrict__ users, const float* __restrict__ contribs,
    const float* __restrict__ cw, const float* __restrict__ cb, const float* __restrict__ calpha,
    const float* __restrict__ urm, const float* __restrict__ urv,
    const float* __restrict__ ubg, const float* __restrict__ ubb,
    const float* __restrict__ ulg, const float* __restrict__ ulb,
    const float* __restrict__ uwv, const float* __restrict__ ubias,
    float* __restrict__ xr, float* __restrict__ cxlast)
{
  __shared__ cpx  sCc[63];
  __shared__ cpx  sUf[64];
  __shared__ cpx  sCa[64];
  __shared__ cpx  sCx[63*64];
  __shared__ cpx  sBuf[8*64];
  __shared__ float sT1[64], sT2[63];
  __shared__ float sU[64];
  __shared__ float sRed[2];

  int b = blockIdx.x, tid = threadIdx.x;

  // 1) DFT63 of contribs (ortho)
  if (tid < 63) {
    cpx acc = cmk(0.f, 0.f);
    for (int j = 0; j < 63; ++j)
      acc = acc + contribs[(size_t)b*63 + j] * cexp_i(-2.f*PI_F*(float)(j*tid)/63.f);
    sCc[tid] = rsqrtf(63.f) * acc;
  }
  // 2) user swnorm
  if (tid < 64) sU[tid] = users[(size_t)b*64 + tid];
  __syncthreads();
  if (tid == 0) {
    float m = 0.f;
    for (int f = 0; f < 64; ++f) m += sU[f];
    m *= (1.f/64.f);
    float v = 0.f;
    for (int f = 0; f < 64; ++f) { float d = sU[f]-m; v += d*d; }
    sRed[0] = m; sRed[1] = v * (1.f/64.f);
  }
  __syncthreads();
  if (tid < 64) {
    float x  = sU[tid];
    float bn = (x - urm[tid]) * rsqrtf(urv[tid] + EPSV) * ubg[tid] + ubb[tid];
    float ln = (x - sRed[0]) * rsqrtf(sRed[1] + EPSV) * ulg[tid] + ulb[tid];
    float w  = sigmoidf_(uwv[tid]);
    sU[tid]  = w*bn + (1.f-w)*ln + ubias[tid];
    sCa[tid] = csigmoid_c(((const cpx*)calpha)[tid]);
  }
  __syncthreads();
  // 3) DFT64 of normalized user (ortho)
  if (tid < 64) {
    cpx acc = cmk(0.f, 0.f);
    for (int f = 0; f < 64; ++f)
      acc = acc + sU[f] * cexp_i(-2.f*PI_F*(float)(f*tid)/64.f);
    sUf[tid] = 0.125f * acc;
  }
  __syncthreads();

  int d = day[b];
  const cpx* cwC = (const cpx*)cw;                      // (63,64,8)
  const cpx* cbC = (const cpx*)cb + (size_t)d*63*64*8;  // (7,63,64,8)

  // 5) fused per-(s,f): ifft8(uf*fft8(c)) == uf*c; combine max|.| and mean
  for (int p = tid; p < 63*64; p += 256) {
    int s = p >> 6, f = p & 63;
    cpx cc = sCc[s], uf = sUf[f];
    const cpx* wrow = cwC + ((size_t)s*64 + f)*8;
    const cpx* brow = cbC + ((size_t)s*64 + f)*8;
    cpx mean = cmk(0.f, 0.f);
    float mx = 0.f;
#pragma unroll
    for (int k = 0; k < 8; ++k) {
      cpx y = uf * (cc * wrow[k] + brow[k]);
      mean = mean + y;
      mx = fmaxf(mx, cabsf_(y));
    }
    mean = 0.125f * mean;
    cpx ca = sCa[f];
    cpx om = cmk(1.f - ca.x, -ca.y);
    sCx[p] = cmk(ca.x*mx, ca.y*mx) + om * mean;
  }
  __syncthreads();

  // 6) 0.1-quantiles: axis1 (over s, n=63, pos=6.2), axis2 (over f, n=64, pos=6.3)
  if (tid < 64) {
    float v6 = kth_abs(&sCx[tid], 63, 64, 6);
    float v7 = kth_abs(&sCx[tid], 63, 64, 7);
    sT1[tid] = v6 + 0.2f*(v7 - v6);
  } else if (tid < 127) {
    int s = tid - 64;
    float v6 = kth_abs(&sCx[s*64], 64, 1, 6);
    float v7 = kth_abs(&sCx[s*64], 64, 1, 7);
    sT2[s] = v6 + 0.3f*(v7 - v6);
  }
  __syncthreads();

  // 7) mask
  for (int p = tid; p < 63*64; p += 256) {
    int s = p >> 6, f = p & 63;
    float a = cabsf_(sCx[p]);
    if (!((a > sT1[f]) && (a > sT2[s]))) sCx[p] = cmk(0.f, 0.f);
  }
  __syncthreads();

  // 8) ifft along f (rows), in-place via 8-row chunks
  for (int s0 = 0; s0 < 63; s0 += 8) {
    int nr = (63 - s0 < 8) ? (63 - s0) : 8;
    for (int i = tid; i < nr*64; i += 256) sBuf[i] = sCx[(s0 + (i >> 6))*64 + (i & 63)];
    __syncthreads();
    for (int i = tid; i < nr*64; i += 256) {
      int r = i >> 6, fp = i & 63;
      cpx acc = cmk(0.f, 0.f);
      for (int f = 0; f < 64; ++f)
        acc = acc + sBuf[r*64 + f] * cexp_i(2.f*PI_F*(float)(f*fp)/64.f);
      sCx[(s0 + r)*64 + fp] = acc;
    }
    __syncthreads();
  }
  // 9) ifft along s (cols), 8-col chunks; ortho scale 1/sqrt(63*64)
  float scale = rsqrtf(63.f * 64.f);
  for (int f0 = 0; f0 < 64; f0 += 8) {
    for (int i = tid; i < 8*63; i += 256) { int c = i/63, s = i%63; sBuf[c*64 + s] = sCx[s*64 + f0 + c]; }
    __syncthreads();
    for (int i = tid; i < 8*63; i += 256) {
      int c = i/63, sp = i%63;
      cpx acc = cmk(0.f, 0.f);
      for (int s = 0; s < 63; ++s)
        acc = acc + sBuf[c*64 + s] * cexp_i(2.f*PI_F*(float)(s*sp)/63.f);
      acc = scale * acc;
      int f = f0 + c;
      xr[(size_t)b*4032 + sp*64 + f] = acc.x;                   // (B,S,F)
      if (sp >= 56)
        ((cpx*)cxlast)[((size_t)b*7 + (sp - 56))*64 + f] = acc; // (B,7,F) complex
    }
    __syncthreads();
  }
}

// =============================================================================
// K2: TDM tile load -> conv1(1x1)+mish -> im2col f16 LDS -> conv2(k=7,s=7).
// Activation tile (63x64 f32, 16KB, contiguous) fetched by the Tensor Data
// Mover into LDS (D# per ISA 8.3/8.4); wave0 issues + waits TENSORcnt, barrier
// publishes. Out-of-range D columns computed-but-discarded => no operand guards.
// =============================================================================
__global__ void __launch_bounds__(256) k_conv12(
    const float* __restrict__ xr, const float* __restrict__ w1, const float* __restrict__ b1,
    const float* __restrict__ w2, const float* __restrict__ b2, float* __restrict__ h2)
{
  __shared__ __attribute__((aligned(64))) float    sxf[64*64];    // [s][f] f32, row 63 unused
  __shared__ __attribute__((aligned(32))) _Float16 sBim[9*1792];  // im2col [n][k=i*7+t]
  int b = blockIdx.x, tid = threadIdx.x;
  int lane = tid & 31, wave = tid >> 5;   // 8 waves
  int n = lane & 15, hh = lane >> 4;

  if (wave == 0) {
    // Tensor DMA descriptor: 1D copy of 4032 dwords global -> LDS
    unsigned long long ga = (unsigned long long)(uintptr_t)(xr + (size_t)b*4032);
    unsigned lds = (unsigned)(uintptr_t)(void*)&sxf[0];   // low 32 bits = LDS byte offset
    u32x4 g0;
    g0[0] = 1u;                                            // count=1 (valid descriptor)
    g0[1] = lds;                                           // lds_addr
    g0[2] = (unsigned)ga;                                  // global_addr[31:0]
    g0[3] = (unsigned)((ga >> 32) & 0x01FFFFFFu) | (2u << 30); // addr[56:32] | type=2
    i32x8 g1;
    g1[0] = (int)(2u << 16);        // workgroup_mask=0, data_size=2 (4B)
    g1[1] = (int)(4032u << 16);     // tensor_dim0 lo16 (=4032)
    g1[2] = (int)(1u << 16);        // tensor_dim0 hi=0, tensor_dim1 lo16=1
    g1[3] = (int)(4032u << 16);     // tensor_dim1 hi=0, tile_dim0=4032
    g1[4] = 1;                      // tile_dim1=1, tile_dim2=0
    g1[5] = 4032;                   // tensor_dim0_stride lo32
    g1[6] = 0;                      // stride hi, tensor_dim1_stride lo
    g1[7] = 0;
    asm volatile("tensor_load_to_lds %0, %1" :: "s"(g0), "s"(g1) : "memory");
    __builtin_amdgcn_s_wait_tensorcnt(0);
  }
  __syncthreads();

  // conv1: 16 Mtiles x 4 Ntiles = 64 tiles over 8 waves; scatter into im2col
  for (int t = wave; t < 64; t += 8) {
    int mt = t >> 2, nt = t & 3;
    int m0 = mt*16, n0 = nt*16;
    v8f acc = {};
#pragma unroll
    for (int kk = 0; kk < 64; kk += 32) {
      v16h a  = frag_a(lane, [&](int m, int k){ return w1[(size_t)(m0+m)*64 + kk + k]; });
      v16h bf = ldB32f(&sxf[(n0 + n)*64 + kk], hh);  // row 63 garbage -> discarded col
      acc = wmma_f16(a, bf, acc);
    }
    int s = n0 + n;
    if (s < 63) {
      int r = s / 7, t7 = s - 7*r;
      _Float16* dst = &sBim[r*1792 + t7];
#pragma unroll
      for (int v = 0; v < 8; ++v) {
        int m = m0 + v + 8*hh;
        dst[m*7] = (_Float16)mishf_(acc[v] + b1[m]);
      }
    }
  }
  __syncthreads();

  // conv2: M=256 (16 tiles), N=9, K=1792 (56 k-steps)
  int r2 = (n < 9) ? n : 8;   // clamped row; cols 9..15 discarded
  for (int mt = wave; mt < 16; mt += 8) {
    int m0 = mt*16;
    v8f acc = {};
#pragma unroll 4
    for (int kk = 0; kk < 1792; kk += 32) {
      __builtin_prefetch(&w2[(size_t)m0*1792 + kk + 512], 0, 1);
      v16h a  = frag_a(lane, [&](int m, int k){ return w2[(size_t)(m0+m)*1792 + kk + k]; });
      v16h bf = ldB16(&sBim[r2*1792 + kk], hh);
      acc = wmma_f16(a, bf, acc);
    }
    if (n < 9) {
#pragma unroll
      for (int v = 0; v < 8; ++v) {
        int m = m0 + v + 8*hh;
        h2[(size_t)b*2304 + m*9 + n] = acc[v] + b2[m];
      }
    }
  }
}

// =============================================================================
// K3b: hid swnorm (bn per channel, ln over 9, sigmoid gate) + mish. Thread/(b,c).
// =============================================================================
__global__ void __launch_bounds__(256) k_hsn(
    const float* __restrict__ h2,
    const float* __restrict__ hrm, const float* __restrict__ hrv,
    const float* __restrict__ hbg, const float* __restrict__ hbb,
    const float* __restrict__ hlg, const float* __restrict__ hlb,
    const float* __restrict__ hwv, const float* __restrict__ hbias,
    float* __restrict__ h2n)
{
  int idx = blockIdx.x*256 + threadIdx.x;
  if (idx >= B_SZ*256) return;
  int c = idx & 255;
  const float* row = h2 + (size_t)idx*9;
  float v[9]; float m = 0.f;
#pragma unroll
  for (int j = 0; j < 9; ++j) { v[j] = row[j]; m += v[j]; }
  m *= (1.f/9.f);
  float var = 0.f;
#pragma unroll
  for (int j = 0; j < 9; ++j) { float dd = v[j]-m; var += dd*dd; }
  var *= (1.f/9.f);
  float w   = sigmoidf_(hwv[c]);
  float irv = rsqrtf(hrv[c] + EPSV), ilv = rsqrtf(var + EPSV);
#pragma unroll
  for (int j = 0; j < 9; ++j) {
    float bn = (v[j] - hrm[c]) * irv * hbg[c] + hbb[c];
    float ln = (v[j] - m) * ilv * hlg[j] + hlb[j];
    h2n[(size_t)idx*9 + j] = mishf_(w*bn + (1.f-w)*ln + hbias[c]);
  }
}

// =============================================================================
// K4: conv3 (k=3): M=64 (4 tiles), N=7, K=768. Block/batch, 128 thr (4 waves).
// =============================================================================
__global__ void __launch_bounds__(128) k_conv3(
    const float* __restrict__ h2n, const float* __restrict__ w3, const float* __restrict__ b3,
    float* __restrict__ h3t)
{
  __shared__ float sh2[2304];
  __shared__ __attribute__((aligned(32))) _Float16 sB3[7*768];
  int b = blockIdx.x, tid = threadIdx.x;
  int lane = tid & 31, wave = tid >> 5;  // 4 waves
  for (int i = tid; i < 2304; i += 128) sh2[i] = h2n[(size_t)b*2304 + i];
  __syncthreads();
  for (int idx = tid; idx < 7*768; idx += 128) {
    int nn = idx / 768, k = idx - nn*768;
    int i = k / 3, t = k - 3*i;
    sB3[idx] = (_Float16)sh2[i*9 + nn + t];
  }
  __syncthreads();
  int n = lane & 15, hh = lane >> 4;
  int r = (n < 7) ? n : 6;   // cols 7..15 discarded
  int m0 = wave*16;
  v8f acc = {};
#pragma unroll 4
  for (int kk = 0; kk < 768; kk += 32) {
    v16h a  = frag_a(lane, [&](int m, int k){ return w3[(size_t)(m0+m)*768 + kk + k]; });
    v16h bf = ldB16(&sB3[r*768 + kk], hh);
    acc = wmma_f16(a, bf, acc);
  }
  if (n < 7) {
#pragma unroll
    for (int v = 0; v < 8; ++v) {
      int m = m0 + v + 8*hh;
      h3t[((size_t)b*7 + n)*64 + m] = acc[v] + b3[m];
    }
  }
}

// =============================================================================
// K5: mid-section. Block/batch, 64 threads.
// =============================================================================
__global__ void __launch_bounds__(64) k_mid(
    const float* __restrict__ h3t, const float* __restrict__ cxlast, const float* __restrict__ halpha,
    const float* __restrict__ frm, const float* __restrict__ frv,
    const float* __restrict__ fbg, const float* __restrict__ fbb,
    const float* __restrict__ flg, const float* __restrict__ flb,
    const float* __restrict__ fwv, const float* __restrict__ fbias,
    float* __restrict__ xs)
{
  __shared__ float sh[7*64];
  __shared__ cpx  sG[7*64];
  __shared__ float sThr[7];
  __shared__ float sX[448];
  __shared__ float sPart[128];
  __shared__ float sMV[2];
  int b = blockIdx.x, tid = threadIdx.x;   // 64 threads

  for (int i = tid; i < 448; i += 64) sh[i] = h3t[(size_t)b*448 + i];
  __syncthreads();

  { // per-column (f' = tid) fft over f, then fft over s, then combine
    cpx col[7];
#pragma unroll
    for (int s = 0; s < 7; ++s) {
      cpx acc = cmk(0.f, 0.f);
      for (int f = 0; f < 64; ++f)
        acc = acc + sh[s*64 + f] * cexp_i(-2.f*PI_F*(float)(f*tid)/64.f);
      col[s] = acc;
    }
    const cpx* cl = (const cpx*)cxlast + (size_t)b*448;  // (7,64)
#pragma unroll
    for (int sp = 0; sp < 7; ++sp) {
      cpx g = cmk(0.f, 0.f), f2 = cmk(0.f, 0.f);
#pragma unroll
      for (int s = 0; s < 7; ++s) {
        cpx tw = cexp_i(-2.f*PI_F*(float)(s*sp)/7.f);
        g  = g  + col[s] * tw;
        f2 = f2 + cl[s*64 + tid] * tw;
      }
      cpx ha = ctanh_c(((const cpx*)halpha)[sp*64 + tid]);
      cpx om = cmk(1.f - ha.x, -ha.y);
      sG[sp*64 + tid] = g*ha + f2*om;
    }
  }
  __syncthreads();
  if (tid < 7) {  // 0.3-quantile over 64: pos = 18.9
    float v18 = kth_abs(&sG[tid*64], 64, 1, 18);
    float v19 = kth_abs(&sG[tid*64], 64, 1, 19);
    sThr[tid] = v18 + 0.9f*(v19 - v18);
  }
  __syncthreads();
  { // mask + 7-pt ifft per column
    cpx col[7];
#pragma unroll
    for (int sp = 0; sp < 7; ++sp) {
      cpx v = sG[sp*64 + tid];
      col[sp] = (cabsf_(v) > sThr[sp]) ? v : cmk(0.f, 0.f);
    }
#pragma unroll
    for (int s = 0; s < 7; ++s) {
      cpx acc = cmk(0.f, 0.f);
#pragma unroll
      for (int sp = 0; sp < 7; ++sp)
        acc = acc + col[sp] * cexp_i(2.f*PI_F*(float)(s*sp)/7.f);
      sX[s*64 + tid] = acc.x * (1.f/7.f);
    }
  }
  __syncthreads();
  { // fin swnorm
    float ssum = 0.f, ssq = 0.f;
    for (int i = tid; i < 448; i += 64) { float x = sX[i]; ssum += x; ssq += x*x; }
    sPart[tid] = ssum; sPart[64 + tid] = ssq;
    __syncthreads();
    if (tid == 0) {
      float S = 0.f, Q = 0.f;
      for (int i = 0; i < 64; ++i) { S += sPart[i]; Q += sPart[64 + i]; }
      float mu = S / 448.f;
      sMV[0] = mu; sMV[1] = Q / 448.f - mu*mu;
    }
    __syncthreads();
    for (int p = tid; p < 448; p += 64) {
      float x  = sX[p];
      float bn = (x - frm[p]) * rsqrtf(frv[p] + EPSV) * fbg[p] + fbb[p];
      float ln = (x - sMV[0]) * rsqrtf(sMV[1] + EPSV) * flg[p] + flb[p];
      float w  = sigmoidf_(fwv[p]);
      xs[(size_t)b*448 + p] = w*bn + (1.f-w)*ln + fbias[p];
    }
  }
}

// =============================================================================
// K6: l1 GEMM across batch: (2048x448) @ (448x256)^T + bias, mish.
// =============================================================================
__global__ void __launch_bounds__(256) k_l1(
    const float* __restrict__ xs, const float* __restrict__ w, const float* __restrict__ bias,
    float* __restrict__ a1)
{
  __shared__ __attribute__((aligned(32))) _Float16 sA[16*448];
  int mb = blockIdx.x, tid = threadIdx.x;
  int lane = tid & 31, wave = tid >> 5;
  int n = lane & 15, hh = lane >> 4;
  for (int i = tid; i < 16*448; i += 256) sA[i] = (_Float16)xs[(size_t)mb*7168 + i];
  __syncthreads();
  for (int nt = wave; nt < 16; nt += 8) {
    int n0 = nt*16;
    v8f acc = {};
#pragma unroll 2
    for (int kk = 0; kk < 448; kk += 32) {
      v16h a  = ldA16(&sA[n*448 + kk], hh);   // A row m = lane&15
      v16h bf = frag_b(lane, [&](int k, int nn){ return w[(size_t)(n0+nn)*448 + kk + k]; });
      acc = wmma_f16(a, bf, acc);
    }
#pragma unroll
    for (int v = 0; v < 8; ++v) {
      int m = v + 8*hh;
      a1[((size_t)mb*16 + m)*256 + n0 + n] = mishf_(acc[v] + bias[n0 + n]);
    }
  }
}

// =============================================================================
// K7: l2 GEMM: (2048x256) @ (256x84)^T + bias -> d_out (B,6,7,2 flattened).
// =============================================================================
__global__ void __launch_bounds__(256) k_l2(
    const float* __restrict__ a1, const float* __restrict__ w, const float* __restrict__ bias,
    float* __restrict__ out)
{
  __shared__ __attribute__((aligned(32))) _Float16 sA[16*256];
  int mb = blockIdx.x, tid = threadIdx.x;
  int lane = tid & 31, wave = tid >> 5;
  int n = lane & 15, hh = lane >> 4;
  for (int i = tid; i < 16*256; i += 256) sA[i] = (_Float16)a1[(size_t)mb*4096 + i];
  __syncthreads();
  if (wave < 6) {  // 84 outputs -> 6 N-tiles (wave-uniform, EXEC stays full)
    int n0 = wave*16;
    int c = n0 + n;
    int cc = (c < 84) ? c : 83;   // clamped; cols >=84 discarded
    v8f acc = {};
#pragma unroll
    for (int kk = 0; kk < 256; kk += 32) {
      v16h a  = ldA16(&sA[n*256 + kk], hh);
      v16h bf = frag_b(lane, [&](int k, int nn){ (void)nn; return w[(size_t)cc*256 + kk + k]; });
      acc = wmma_f16(a, bf, acc);
    }
    if (c < 84) {
#pragma unroll
      for (int v = 0; v < 8; ++v) {
        int m = v + 8*hh;
        out[((size_t)mb*16 + m)*84 + c] = acc[v] + bias[c];
      }
    }
  }
}

// ---------------- input indices (setup_inputs dict order, params flattened) ----
enum {
  IN_DAY = 0, IN_USERS, IN_CONTRIBS,
  P_CW, P_CB, P_CALPHA, P_HALPHA,
  U_RM, U_RV, U_BG, U_BB, U_LG, U_LB, U_W, U_B,
  H_RM, H_RV, H_BG, H_BB, H_LG, H_LB, H_W, H_B,
  F_RM, F_RV, F_BG, F_BB, F_LG, F_LB, F_W, F_B,
  P_C1W, P_C1B, P_C2W, P_C2B, P_C3W, P_C3B,
  P_L1W, P_L1B, P_L2W, P_L2B
};

extern "C" void kernel_launch(void* const* d_in, const int* in_sizes, int n_in,
                              void* d_out, int out_size, void* d_ws, size_t ws_size,
                              hipStream_t stream) {
  (void)in_sizes; (void)n_in; (void)out_size; (void)ws_size;
  const int*   day      = (const int*)  d_in[IN_DAY];
  const float* users    = (const float*)d_in[IN_USERS];
  const float* contribs = (const float*)d_in[IN_CONTRIBS];

  float* ws = (float*)d_ws;
  size_t o = 0;
  float* xr  = ws + o; o += (size_t)B_SZ*4032;    // conv1 input (B,S,F)
  float* cxl = ws + o; o += (size_t)B_SZ*7*64*2;  // complex last rows (B,7,F)
  float* h2  = ws + o; o += (size_t)B_SZ*2304;    // conv2 out (B,256,9)
  float* h2n = ws + o; o += (size_t)B_SZ*2304;    // after swnorm+mish
  float* h3t = ws + o; o += (size_t)B_SZ*448;     // conv3 out transposed (B,7,64)
  float* xs  = ws + o; o += (size_t)B_SZ*448;     // after fin swnorm
  float* a1  = ws + o; o += (size_t)B_SZ*256;     // l1 activations

  k_frontend<<<B_SZ, 256, 0, stream>>>(
      day, users, contribs,
      (const float*)d_in[P_CW], (const float*)d_in[P_CB], (const float*)d_in[P_CALPHA],
      (const float*)d_in[U_RM], (const float*)d_in[U_RV],
      (const float*)d_in[U_BG], (const float*)d_in[U_BB],
      (const float*)d_in[U_LG], (const float*)d_in[U_LB],
      (const float*)d_in[U_W],  (const float*)d_in[U_B],
      xr, cxl);

  k_conv12<<<B_SZ, 256, 0, stream>>>(
      xr, (const float*)d_in[P_C1W], (const float*)d_in[P_C1B],
      (const float*)d_in[P_C2W], (const float*)d_in[P_C2B], h2);

  k_hsn<<<B_SZ, 256, 0, stream>>>(
      h2,
      (const float*)d_in[H_RM], (const float*)d_in[H_RV],
      (const float*)d_in[H_BG], (const float*)d_in[H_BB],
      (const float*)d_in[H_LG], (const float*)d_in[H_LB],
      (const float*)d_in[H_W],  (const float*)d_in[H_B],
      h2n);

  k_conv3<<<B_SZ, 128, 0, stream>>>(
      h2n, (const float*)d_in[P_C3W], (const float*)d_in[P_C3B], h3t);

  k_mid<<<B_SZ, 64, 0, stream>>>(
      h3t, cxl, (const float*)d_in[P_HALPHA],
      (const float*)d_in[F_RM], (const float*)d_in[F_RV],
      (const float*)d_in[F_BG], (const float*)d_in[F_BB],
      (const float*)d_in[F_LG], (const float*)d_in[F_LB],
      (const float*)d_in[F_W],  (const float*)d_in[F_B],
      xs);

  k_l1<<<B_SZ/16, 256, 0, stream>>>(
      xs, (const float*)d_in[P_L1W], (const float*)d_in[P_L1B], a1);

  k_l2<<<B_SZ/16, 256, 0, stream>>>(
      a1, (const float*)d_in[P_L2W], (const float*)d_in[P_L2B], (float*)d_out);
}
